// OptimizedEdgeEnhancement_22419729285389
// MI455X (gfx1250) — matrix operand order repaired and checked by reading, not compiled, not run on hardware
//
#include <hip/hip_runtime.h>
#include <hip/hip_bf16.h>

typedef __attribute__((ext_vector_type(16))) _Float16 v16h;
typedef __attribute__((ext_vector_type(8)))  _Float16 v8h;
typedef __attribute__((ext_vector_type(8)))  float    v8f;

#define HW   65536   // 256*256
#define CIN  256
#define CMID 64
#define NB   4

// ---------------------------------------------------------------------------
// Kernel 0: zero the small reduction scratch (max + per-plane sums)
// ---------------------------------------------------------------------------
__global__ void init_scratch(float* smallbuf, int n) {
    int i = blockIdx.x * blockDim.x + threadIdx.x;
    if (i < n) smallbuf[i] = 0.0f;
}

// ---------------------------------------------------------------------------
// Kernel 1: x [B,256,H,W] --(w_in 64x256, b_in)--> x1 [B,64,H,W]
// GEMM per batch: D[o,p] = sum_c W[o,c] * X[c,p],  M=64, K=256, N=65536
// Block: 256 thr = 8 waves; wave computes 16(o) x 16(p); chTile=wave&3, pixTile=wave>>2
// LDS B-tile stored pixel-major so each lane's 16 K-halfs are contiguous
// -> B fragment = 2x ds_load_b128 (same as A), no scalar u16 gathers.
// ---------------------------------------------------------------------------
__global__ __launch_bounds__(256) void proj_gemm(
    const float* __restrict__ x, const float* __restrict__ w_in,
    const float* __restrict__ b_in, float* __restrict__ x1)
{
    __shared__ _Float16 wlds[CMID * CIN];   // 32 KB: all weights as f16
    __shared__ _Float16 xtileT[32][40];     // [pixel][channel]; 80B rows, 16B-aligned

    const int tid  = threadIdx.x;
    const int lane = tid & 31;
    const int wave = tid >> 5;
    const int b    = blockIdx.y;
    const int p0blk = blockIdx.x * 32;

    for (int i = tid; i < CMID * CIN; i += 256)
        wlds[i] = (_Float16)w_in[i];
    __syncthreads();

    const int  chTile  = wave & 3;
    const int  pixTile = wave >> 2;
    const int  laneLow = lane & 15;
    const bool hi      = lane >= 16;
    const int  p0      = p0blk + pixTile * 16;

    const float* xb = x + (size_t)b * CIN * HW;
    v8f acc = {};

    for (int k0 = 0; k0 < CIN; k0 += 32) {
        __syncthreads();
        {   // cooperative stage with transpose: 32 channels x 32 pixels, f32 -> f16
            int r  = tid >> 3;          // 0..31  (channel row)
            int cq = (tid & 7) * 4;     // 0..28  (pixel quad)
            const float4 v = *(const float4*)(xb + (size_t)(k0 + r) * HW + p0blk + cq);
            xtileT[cq + 0][r] = (_Float16)v.x;
            xtileT[cq + 1][r] = (_Float16)v.y;
            xtileT[cq + 2][r] = (_Float16)v.z;
            xtileT[cq + 3][r] = (_Float16)v.w;
        }
        __syncthreads();

        // A fragment (16x32, f16): lanes<16 hold K 0..15, lanes>=16 K 16..31
        v16h a;
        {
            const _Float16* wrow = &wlds[(chTile * 16 + laneLow) * CIN + k0];
            int kb = hi ? 16 : 0;
            #pragma unroll
            for (int i = 0; i < 8; ++i) a[i]     = wrow[kb + i];
            #pragma unroll
            for (int i = 0; i < 8; ++i) a[8 + i] = wrow[8 + kb + i];
        }
        // B fragment (32x16, f16): lanes<16 -> K 0..15, lanes>=16 -> K 16..31; N = lane%16
        v16h bf;
        {
            int kb = hi ? 16 : 0;
            int n  = pixTile * 16 + laneLow;
            const v8h* bp = (const v8h*)&xtileT[n][kb];   // 16B-aligned
            v8h b0 = bp[0];
            v8h b1 = bp[1];
            #pragma unroll
            for (int i = 0; i < 8; ++i) { bf[i] = b0[i]; bf[8 + i] = b1[i]; }
        }
        acc = __builtin_amdgcn_wmma_f32_16x16x32_f16(false, a, false, bf,
                                                     (short)0, acc, false, false);
    }

    // C/D layout: VGPR r, lanes<16 -> M=r, lanes>=16 -> M=r+8; N = lane%16
    #pragma unroll
    for (int r = 0; r < 8; ++r) {
        int oc  = chTile * 16 + r + (hi ? 8 : 0);
        int pix = p0 + laneLow;
        x1[((size_t)b * CMID + oc) * HW + pix] = acc[r] + b_in[oc];
    }
}

// ---------------------------------------------------------------------------
// Kernel 2: 5x5 Gaussian depthwise (zero padding), separable weights
// ---------------------------------------------------------------------------
__global__ __launch_bounds__(256) void gauss5(const float* __restrict__ src,
                                              float* __restrict__ dst)
{
    const int idx   = blockIdx.x * 256 + threadIdx.x;   // 16.7M
    const int plane = idx >> 16;
    const int pix   = idx & 65535;
    const int h     = pix >> 8;
    const int w     = pix & 255;
    const float* p  = src + (size_t)plane * HW;

    const float g1d[5] = {0.6065307f, 0.8824969f, 1.0f, 0.8824969f, 0.6065307f};
    const float inv    = 1.0f / 15.824923f;

    __builtin_prefetch(p + pix + 512, 0, 0);  // global_prefetch_b8 two rows ahead

    float acc = 0.0f;
    #pragma unroll
    for (int dy = 0; dy < 5; ++dy) {
        int hh = h + dy - 2;
        if ((unsigned)hh >= 256u) continue;
        float row = 0.0f;
        #pragma unroll
        for (int dx = 0; dx < 5; ++dx) {
            int ww = w + dx - 2;
            float v = ((unsigned)ww < 256u) ? p[hh * 256 + ww] : 0.0f;
            row += g1d[dx] * v;
        }
        acc += g1d[dy] * row;
    }
    dst[idx] = acc * inv;
}

// ---------------------------------------------------------------------------
// Kernel 3: fused Scharr0/Scharr90/Laplacian + edge combine
//           + block reductions: global max (atomic on uint bits, e>=0)
//           and per-(b,c) sum (atomicAdd) for the SE pooling
// ---------------------------------------------------------------------------
__global__ __launch_bounds__(256) void edge_k(const float* __restrict__ sm,
                                              float* __restrict__ edge,
                                              float* __restrict__ maxv,
                                              float* __restrict__ psum)
{
    const int tid   = threadIdx.x;
    const int idx   = blockIdx.x * 256 + tid;
    const int plane = idx >> 16;
    const int pix   = idx & 65535;
    const int h     = pix >> 8;
    const int w     = pix & 255;
    const float* p  = sm + (size_t)plane * HW;

    float v[3][3];
    #pragma unroll
    for (int dy = 0; dy < 3; ++dy) {
        int hh = h + dy - 1;
        #pragma unroll
        for (int dx = 0; dx < 3; ++dx) {
            int ww = w + dx - 1;
            v[dy][dx] = ((unsigned)hh < 256u && (unsigned)ww < 256u)
                          ? p[hh * 256 + ww] : 0.0f;
        }
    }
    float g0  = 3.0f * (v[0][0] - v[0][2]) + 10.0f * (v[1][0] - v[1][2])
              + 3.0f * (v[2][0] - v[2][2]);
    float g90 = 3.0f * (v[0][0] - v[2][0]) + 10.0f * (v[0][1] - v[2][1])
              + 3.0f * (v[0][2] - v[2][2]);
    float lap = 4.0f * v[1][1] - v[0][1] - v[1][0] - v[1][2] - v[2][1];
    float e   = fmaxf(fabsf(g0), fabsf(g90)) + 0.1f * fabsf(lap);
    edge[idx] = e;

    __shared__ float smax[256];
    __shared__ float ssum[256];
    smax[tid] = e;
    ssum[tid] = e;
    __syncthreads();
    for (int s = 128; s > 0; s >>= 1) {
        if (tid < s) {
            smax[tid] = fmaxf(smax[tid], smax[tid + s]);
            ssum[tid] += ssum[tid + s];
        }
        __syncthreads();
    }
    if (tid == 0) {
        atomicMax((unsigned int*)maxv, __float_as_uint(smax[0]));  // e >= 0
        atomicAdd(&psum[plane], ssum[0]);
    }
}

// ---------------------------------------------------------------------------
// Kernel 4: SE head. pooled_mean = psum / (HW*(max+eps)); clip is identity.
// sfold[b][c] = sigmoid(fc2(relu(fc1(pooled)))) / (max+eps)
// ---------------------------------------------------------------------------
__global__ __launch_bounds__(256) void se_k(const float* __restrict__ psum,
                                            const float* __restrict__ maxv,
                                            const float* __restrict__ w_fc1,
                                            const float* __restrict__ b_fc1,
                                            const float* __restrict__ w_fc2,
                                            const float* __restrict__ b_fc2,
                                            float* __restrict__ sfold)
{
    const int t = threadIdx.x;          // 256 = 4 batches x 64 channels
    const int b = t >> 6;
    const int o = t & 63;
    const float denom = maxv[0] + 1e-8f;
    const float inv   = 1.0f / (65536.0f * denom);

    float hj[4];
    #pragma unroll
    for (int j = 0; j < 4; ++j) {
        float s = b_fc1[j];
        for (int c = 0; c < 64; ++c)
            s += w_fc1[j * 64 + c] * (psum[b * 64 + c] * inv);
        hj[j] = fmaxf(s, 0.0f);
    }
    float s2 = b_fc2[o];
    #pragma unroll
    for (int j = 0; j < 4; ++j) s2 += w_fc2[o * 4 + j] * hj[j];
    float sc = 1.0f / (1.0f + __expf(-s2));
    sfold[b * 64 + o] = sc / denom;     // folds normalization + SE scale
}

// ---------------------------------------------------------------------------
// Kernel 5: out[b,o,p] = sum_c w_out[o,c] * (edge[b,c,p]*sfold[b,c]) + b_out[o]
// WMMA GEMM M=64, K=64, N=65536; sfold applied while staging B into LDS.
// Same pixel-major LDS layout as proj_gemm.
// ---------------------------------------------------------------------------
__global__ __launch_bounds__(256) void out_gemm(
    const float* __restrict__ edge, const float* __restrict__ w_out,
    const float* __restrict__ b_out, const float* __restrict__ sfold,
    float* __restrict__ out)
{
    __shared__ _Float16 wlds[CMID * CMID];  // 8 KB
    __shared__ _Float16 xtileT[32][40];     // [pixel][channel]
    __shared__ float    sf[CMID];

    const int tid  = threadIdx.x;
    const int lane = tid & 31;
    const int wave = tid >> 5;
    const int b    = blockIdx.y;
    const int p0blk = blockIdx.x * 32;

    for (int i = tid; i < CMID * CMID; i += 256)
        wlds[i] = (_Float16)w_out[i];
    if (tid < CMID) sf[tid] = sfold[b * CMID + tid];
    __syncthreads();

    const int  chTile  = wave & 3;
    const int  pixTile = wave >> 2;
    const int  laneLow = lane & 15;
    const bool hi      = lane >= 16;
    const int  p0      = p0blk + pixTile * 16;

    const float* eb = edge + (size_t)b * CMID * HW;
    v8f acc = {};

    for (int k0 = 0; k0 < CMID; k0 += 32) {
        __syncthreads();
        {
            int r  = tid >> 3;
            int cq = (tid & 7) * 4;
            const float4 v = *(const float4*)(eb + (size_t)(k0 + r) * HW + p0blk + cq);
            const float s = sf[k0 + r];
            xtileT[cq + 0][r] = (_Float16)(v.x * s);
            xtileT[cq + 1][r] = (_Float16)(v.y * s);
            xtileT[cq + 2][r] = (_Float16)(v.z * s);
            xtileT[cq + 3][r] = (_Float16)(v.w * s);
        }
        __syncthreads();

        v16h a;
        {
            const _Float16* wrow = &wlds[(chTile * 16 + laneLow) * CMID + k0];
            int kb = hi ? 16 : 0;
            #pragma unroll
            for (int i = 0; i < 8; ++i) a[i]     = wrow[kb + i];
            #pragma unroll
            for (int i = 0; i < 8; ++i) a[8 + i] = wrow[8 + kb + i];
        }
        v16h bf;
        {
            int kb = hi ? 16 : 0;
            int n  = pixTile * 16 + laneLow;
            const v8h* bp = (const v8h*)&xtileT[n][kb];
            v8h b0 = bp[0];
            v8h b1 = bp[1];
            #pragma unroll
            for (int i = 0; i < 8; ++i) { bf[i] = b0[i]; bf[8 + i] = b1[i]; }
        }
        acc = __builtin_amdgcn_wmma_f32_16x16x32_f16(false, a, false, bf,
                                                     (short)0, acc, false, false);
    }

    #pragma unroll
    for (int r = 0; r < 8; ++r) {
        int oc  = chTile * 16 + r + (hi ? 8 : 0);
        int pix = p0 + laneLow;
        out[((size_t)b * CMID + oc) * HW + pix] = acc[r] + b_out[oc];
    }
}

// ---------------------------------------------------------------------------
extern "C" void kernel_launch(void* const* d_in, const int* in_sizes, int n_in,
                              void* d_out, int out_size, void* d_ws, size_t ws_size,
                              hipStream_t stream)
{
    const float* x     = (const float*)d_in[0];
    const float* w_in  = (const float*)d_in[1];
    const float* b_in  = (const float*)d_in[2];
    const float* w_fc1 = (const float*)d_in[3];
    const float* b_fc1 = (const float*)d_in[4];
    const float* w_fc2 = (const float*)d_in[5];
    const float* b_fc2 = (const float*)d_in[6];
    const float* w_out = (const float*)d_in[7];
    const float* b_out = (const float*)d_in[8];
    float* out = (float*)d_out;

    const size_t planeElems = (size_t)NB * CMID * HW;   // 16,777,216
    float* buf0  = (float*)d_ws;                 // x1, later reused as edge
    float* buf1  = buf0 + planeElems;            // sm (gaussian output)
    float* smallb = buf1 + planeElems;           // [0]=max, [1..256]=psum, [257..512]=sfold
    float* maxv  = smallb;
    float* psum  = smallb + 1;
    float* sfold = smallb + 1 + NB * CMID;

    // 0) zero reduction scratch
    init_scratch<<<2, 256, 0, stream>>>(smallb, 1 + 2 * NB * CMID);

    // 1) input projection GEMM (WMMA):  x -> buf0
    proj_gemm<<<dim3(HW / 32, NB), 256, 0, stream>>>(x, w_in, b_in, buf0);

    // 2) gaussian blur: buf0 -> buf1
    gauss5<<<(int)(planeElems / 256), 256, 0, stream>>>(buf0, buf1);

    // 3) fused edge + reductions: buf1 -> buf0 (edge), maxv, psum
    edge_k<<<(int)(planeElems / 256), 256, 0, stream>>>(buf1, buf0, maxv, psum);

    // 4) SE head -> folded per-(b,c) scale
    se_k<<<1, 256, 0, stream>>>(psum, maxv, w_fc1, b_fc1, w_fc2, b_fc2, sfold);

    // 5) output projection GEMM (WMMA) with folded scale: buf0 -> out
    out_gemm<<<dim3(HW / 32, NB), 256, 0, stream>>>(buf0, w_out, b_out, sfold, out);
}
